// NodeFeatureNet_45165876085118
// MI455X (gfx1250) — compile-verified, matrix operand order checked by reading, not run on hardware
//
#include <hip/hip_runtime.h>
#include <hip/hip_bf16.h>
#include <stdint.h>

// ---------------------------------------------------------------------------
// CDNA5 (gfx1250) implementation: bf16 WMMA GEMM chain + gather/scatter glue.
// ---------------------------------------------------------------------------

typedef __attribute__((ext_vector_type(16))) __bf16 v16bf;
typedef __attribute__((ext_vector_type(8)))  __bf16 v8bf;
typedef __attribute__((ext_vector_type(8)))  float  v8f;
typedef __attribute__((ext_vector_type(4)))  int    v4i;

#define NEG_SLOPE 0.2f

__device__ __forceinline__ float leaky(float x) {
    return x >= 0.0f ? x : NEG_SLOPE * x;
}

// Async global->LDS staging (GLOBAL_LOAD_ASYNC_TO_LDS_B128, ASYNCcnt-tracked)
#if defined(__has_builtin)
#if __has_builtin(__builtin_amdgcn_global_load_async_to_lds_b128)
#define HAVE_ASYNC_LDS 1
#endif
#endif
#ifndef HAVE_ASYNC_LDS
#define HAVE_ASYNC_LDS 0
#endif

// builtin expects (addrspace(1) v4i*, addrspace(3) v4i*, int offset, int cpol)
typedef __attribute__((address_space(1))) v4i as1_v4i;
typedef __attribute__((address_space(3))) v4i as3_v4i;

__device__ __forceinline__ void wait_async_zero() {
#if __has_builtin(__builtin_amdgcn_s_wait_asynccnt)
    __builtin_amdgcn_s_wait_asynccnt(0);
#else
    asm volatile("s_wait_asynccnt 0" ::: "memory");
#endif
}

// ------------------------- utility kernels ---------------------------------

__global__ void zero_f32_kernel(float* __restrict__ p, int n) {
    int t = blockIdx.x * blockDim.x + threadIdx.x;
    if (t < n) p[t] = 0.0f;
}

// Convert f32 weight [N, Ksrc] row-major -> bf16 [N, Kp] row-major, zero-padded K.
__global__ void cvt_weight_kernel(const float* __restrict__ src, __bf16* __restrict__ dst,
                                  int N, int Ksrc, int Kp) {
    int t = blockIdx.x * blockDim.x + threadIdx.x;
    if (t >= N * Kp) return;
    int n = t / Kp;
    int k = t - n * Kp;
    float v = (k < Ksrc) ? src[(unsigned)(n * Ksrc + k)] : 0.0f;
    dst[(unsigned)t] = (__bf16)v;
}

// ------------------------- face-normal scatter -----------------------------

__global__ void face_normal_kernel(const float* __restrict__ v, const int* __restrict__ f,
                                   float* __restrict__ vn, int NF) {
    int t = blockIdx.x * blockDim.x + threadIdx.x;
    if (t >= NF) return;
    int i0 = f[3 * t + 0], i1 = f[3 * t + 1], i2 = f[3 * t + 2];
    float p0x = v[3 * i0], p0y = v[3 * i0 + 1], p0z = v[3 * i0 + 2];
    float p1x = v[3 * i1], p1y = v[3 * i1 + 1], p1z = v[3 * i1 + 2];
    float p2x = v[3 * i2], p2y = v[3 * i2 + 1], p2z = v[3 * i2 + 2];
    float e1x = p1x - p0x, e1y = p1y - p0y, e1z = p1z - p0z;
    float e2x = p2x - p0x, e2y = p2y - p0y, e2z = p2z - p0z;
    float nx = e1y * e2z - e1z * e2y;
    float ny = e1z * e2x - e1x * e2z;
    float nz = e1x * e2y - e1y * e2x;
    atomicAdd(&vn[3 * i0 + 0], nx); atomicAdd(&vn[3 * i0 + 1], ny); atomicAdd(&vn[3 * i0 + 2], nz);
    atomicAdd(&vn[3 * i1 + 0], nx); atomicAdd(&vn[3 * i1 + 1], ny); atomicAdd(&vn[3 * i1 + 2], nz);
    atomicAdd(&vn[3 * i2 + 0], nx); atomicAdd(&vn[3 * i2 + 1], ny); atomicAdd(&vn[3 * i2 + 2], nz);
}

// --------------- per-vertex: trilinear cube sampling + z_point -------------
// One 128-thread block per (padded) vertex. Thread j computes:
//   - z_point channel j (W1: [128,6]) -> zcat[i*256 + j]   (bf16)
//   - trilinear sample j (j<125)      -> cubes[i*128 + j]  (bf16, 125..127 = 0)
// Rows i >= M are zero-filled so the GEMM chain needs no row guards.

__global__ void vertex_kernel(const float* __restrict__ v, const float* __restrict__ vnacc,
                              const float* __restrict__ W1, const float* __restrict__ b1,
                              const float* __restrict__ vol,
                              __bf16* __restrict__ zcat, __bf16* __restrict__ cubes,
                              int M) {
    const int i = blockIdx.x;
    const int j = threadIdx.x;   // 0..127

    if (i >= M) {                // zero-pad tail rows (Mp - M of them)
        cubes[(unsigned)i * 128u + j] = (__bf16)0.0f;
        zcat[(unsigned)i * 256u + j] = (__bf16)0.0f;
        return;
    }

    float vx = v[3 * i], vy = v[3 * i + 1], vz = v[3 * i + 2];
    float nx = vnacc[3 * i], ny = vnacc[3 * i + 1], nz = vnacc[3 * i + 2];
    float nrm = sqrtf(nx * nx + ny * ny + nz * nz);
    nrm = fmaxf(nrm, 1e-6f);
    nx /= nrm; ny /= nrm; nz /= nrm;

    // z_point = leaky([v, vn] @ W1^T + b1), channel j
    float zp = b1[j];
    zp += vx * W1[j * 6 + 0];
    zp += vy * W1[j * 6 + 1];
    zp += vz * W1[j * 6 + 2];
    zp += nx * W1[j * 6 + 3];
    zp += ny * W1[j * 6 + 4];
    zp += nz * W1[j * 6 + 5];
    zcat[(unsigned)i * 256u + j] = (__bf16)leaky(zp);

    // trilinear sample
    float smp = 0.0f;
    if (j < 125) {
        // x_shift[idx] = (grid[idx/25], grid[(idx/5)%5], grid[idx%5]),
        // grid = linspace(-3, 2, 5) = -3 + 1.25*t   (note: -K//2 == -3 in Python)
        int ja = j / 25;
        int jb = (j / 5) % 5;
        int jc = j % 5;
        const float sc = 2.0f / 192.0f;   // /Dm * 2
        float gx = vx + (-3.0f + 1.25f * (float)ja) * sc;
        float gy = vy + (-3.0f + 1.25f * (float)jb) * sc;
        float gz = vz + (-3.0f + 1.25f * (float)jc) * sc;

        float ix = fminf(fmaxf((gx + 1.0f) * 0.5f * 191.0f, 0.0f), 191.0f);
        float iy = fminf(fmaxf((gy + 1.0f) * 0.5f * 191.0f, 0.0f), 191.0f);
        float iz = fminf(fmaxf((gz + 1.0f) * 0.5f * 191.0f, 0.0f), 191.0f);
        float fx = floorf(ix), fy = floorf(iy), fz = floorf(iz);
        int x0 = (int)fx, y0 = (int)fy, z0 = (int)fz;
        int x1 = min(x0 + 1, 191), y1 = min(y0 + 1, 191), z1 = min(z0 + 1, 191);
        float wx = ix - fx, wy = iy - fy, wz = iz - fz;

#define VAT(zz, yy, xx) vol[(unsigned)(((zz) * 192 + (yy)) * 192 + (xx))]
        float c000 = VAT(z0, y0, x0), c001 = VAT(z0, y0, x1);
        float c010 = VAT(z0, y1, x0), c011 = VAT(z0, y1, x1);
        float c100 = VAT(z1, y0, x0), c101 = VAT(z1, y0, x1);
        float c110 = VAT(z1, y1, x0), c111 = VAT(z1, y1, x1);
#undef VAT
        float c00 = c000 * (1.0f - wx) + c001 * wx;
        float c01 = c010 * (1.0f - wx) + c011 * wx;
        float c10 = c100 * (1.0f - wx) + c101 * wx;
        float c11 = c110 * (1.0f - wx) + c111 * wx;
        float c0 = c00 * (1.0f - wy) + c01 * wy;
        float c1 = c10 * (1.0f - wy) + c11 * wy;
        smp = c0 * (1.0f - wz) + c1 * wz;
    }
    cubes[(unsigned)i * 128u + j] = (__bf16)smp;
}

// ------------------------- WMMA bf16 GEMM ----------------------------------
// D[M,N] = leaky(A[M,K] @ B[N,K]^T + bias), A row-major bf16 (rows padded to
// a multiple of 64 and fully defined), B row-major [N][K] bf16 (== column-
// major [K][N], exactly the WMMA B layout, same per-lane addressing as A).
// Workgroup = 256 threads = 8 waves; 64 M-rows staged in LDS via async
// global->LDS copies; wave w owns M-subtile (w&3) and N-half (w>>2), and
// processes two 16-col tiles concurrently (two independent wmma streams
// sharing one A fragment -> B loads of one tile overlap wmma of the other).
// All global offsets kept in 32-bit so the backend can use SGPR-base +
// 32-bit-voffset addressing instead of per-lane 64-bit pointer adds.

#define GEMM_ROWS 64

// fused bias + leaky + store of one 16x16 D tile
template <bool OUT_F32, bool GUARD>
__device__ __forceinline__ void store_tile(
        const v8f& acc, const float* __restrict__ bias, void* __restrict__ out,
        int Mtot, int ldout, int col_off, int m_base, int m0, int kh, int col, int n0) {
    const float bn = bias[(unsigned)(n0 + col)];
    const unsigned obase = (unsigned)(m_base + m0 + 8 * kh) * (unsigned)ldout
                         + (unsigned)(col_off + n0 + col);
    if (!GUARD || (m_base + m0 + 16 <= Mtot)) {
        // full tile: no per-lane predication (scalar-uniform fast path)
#pragma unroll
        for (int r = 0; r < 8; ++r) {
            float val = leaky(acc[r] + bn);
            unsigned off = obase + (unsigned)r * (unsigned)ldout;
            if (OUT_F32) __builtin_nontemporal_store(val, (float*)out + off);
            else         ((__bf16*)out)[off] = (__bf16)val;
        }
    } else {
        // partial tile at the bottom edge (rare): per-row guard
#pragma unroll
        for (int r = 0; r < 8; ++r) {
            int m = m_base + m0 + r + 8 * kh;
            if (m < Mtot) {
                float val = leaky(acc[r] + bn);
                unsigned off = (unsigned)m * (unsigned)ldout
                             + (unsigned)(col_off + n0 + col);
                if (OUT_F32) __builtin_nontemporal_store(val, (float*)out + off);
                else         ((__bf16*)out)[off] = (__bf16)val;
            }
        }
    }
}

template <int K, bool OUT_F32, bool GUARD>
__global__ __launch_bounds__(256) void gemm_bf16_kernel(
        const __bf16* __restrict__ A, const __bf16* __restrict__ B,
        const float* __restrict__ bias, void* __restrict__ out,
        int Mtot, int N, int ldout, int col_off) {
    __shared__ __bf16 sA[GEMM_ROWS * K];

    const int tid = threadIdx.x;
    const int m_base = blockIdx.x * GEMM_ROWS;

    // ---- stage contiguous 64 x K bf16 A panel into LDS ----
    constexpr int TOTAL = GEMM_ROWS * K / 8;   // 16-byte chunks
    const unsigned ablk = (unsigned)m_base * (unsigned)K;   // element offset
#if HAVE_ASYNC_LDS
    {
        as1_v4i* gsrc = (as1_v4i*)(A + ablk);
        as3_v4i* ldst = (as3_v4i*)&sA[0];
#pragma unroll 4
        for (int c = tid; c < TOTAL; c += 256) {
            __builtin_amdgcn_global_load_async_to_lds_b128(gsrc + c, ldst + c, 0, 0);
        }
        wait_async_zero();
    }
#else
    for (int c = tid; c < TOTAL; c += 256) {
        *(v8bf*)(&sA[c * 8]) = *(const v8bf*)(A + ablk + (unsigned)(c * 8));
    }
#endif
    __syncthreads();

    const int lane = tid & 31;
    const int wave = tid >> 5;
    const int col = lane & 15;     // N index within tile / M row within A frag
    const int kh  = lane >> 4;     // K-half select per 16-bit WMMA layout
    const int m0  = (wave & 3) * 16;
    const int nhalf = wave >> 2;   // 0..1
    const int ntiles = N >> 5;     // tiles per wave (N/2/16), always even here

    // warm the weight panel into L2
    __builtin_prefetch(B + (unsigned)(nhalf * (N >> 1)) * (unsigned)K, 0, 3);

    for (int t = 0; t < ntiles; t += 2) {
        const int n0 = nhalf * (N >> 1) + t * 16;
        const unsigned brow0 = (unsigned)(n0 + col) * (unsigned)K + (unsigned)(kh * 8);
        const unsigned brow1 = brow0 + 16u * (unsigned)K;
        v8f acc0 = {}, acc1 = {};
#pragma unroll
        for (int k0 = 0; k0 < K; k0 += 32) {
            // A fragment (shared by both tiles): row = lane&15;
            // lane<16 -> K = k0+[0..7], k0+[16..23]; lane>=16 -> +8 ranges
            const __bf16* ap = &sA[(m0 + col) * K + k0 + kh * 8];
            v8bf alo = *(const v8bf*)ap;
            v8bf ahi = *(const v8bf*)(ap + 16);
            // B fragments: same addressing against rows n0+col / n0+16+col
            v8bf b0lo = *(const v8bf*)(B + brow0 + (unsigned)k0);
            v8bf b0hi = *(const v8bf*)(B + brow0 + (unsigned)(k0 + 16));
            v8bf b1lo = *(const v8bf*)(B + brow1 + (unsigned)k0);
            v8bf b1hi = *(const v8bf*)(B + brow1 + (unsigned)(k0 + 16));
            v16bf av, bv0, bv1;
#pragma unroll
            for (int q = 0; q < 8; ++q) {
                av[q]  = alo[q];  av[q + 8]  = ahi[q];
                bv0[q] = b0lo[q]; bv0[q + 8] = b0hi[q];
                bv1[q] = b1lo[q]; bv1[q + 8] = b1hi[q];
            }
            acc0 = __builtin_amdgcn_wmma_f32_16x16x32_bf16(
                false, av, false, bv0, (short)0, acc0, false, false);
            acc1 = __builtin_amdgcn_wmma_f32_16x16x32_bf16(
                false, av, false, bv1, (short)0, acc1, false, false);
        }
        store_tile<OUT_F32, GUARD>(acc0, bias, out, Mtot, ldout, col_off,
                                   m_base, m0, kh, col, n0);
        store_tile<OUT_F32, GUARD>(acc1, bias, out, Mtot, ldout, col_off,
                                   m_base, m0, kh, col, n0 + 16);
    }
}

// ------------------------------ launch -------------------------------------

extern "C" void kernel_launch(void* const* d_in, const int* in_sizes, int n_in,
                              void* d_out, int out_size, void* d_ws, size_t ws_size,
                              hipStream_t stream) {
    const float* v     = (const float*)d_in[0];   // [1,M,3]
    const int*   f     = (const int*)  d_in[1];   // [1,NF,3]
    const float* vol   = (const float*)d_in[2];   // [1,1,192,192,192]
    const float* Wconv = (const float*)d_in[3];   // [128,1,5,5,5] -> [128,125]
    const float* bconv = (const float*)d_in[4];
    const float* Wlfc  = (const float*)d_in[5];   // [128,128]
    const float* blfc  = (const float*)d_in[6];
    const float* W1    = (const float*)d_in[7];   // [128,6]
    const float* b1    = (const float*)d_in[8];
    const float* W2    = (const float*)d_in[9];   // [512,256]
    const float* b2    = (const float*)d_in[10];
    const float* W3    = (const float*)d_in[11];  // [256,512]
    const float* b3    = (const float*)d_in[12];
    float* out = (float*)d_out;                   // [1,M,256]

    const int M  = in_sizes[0] / 3;
    const int NF = in_sizes[1] / 3;
    const int mblocks = (M + GEMM_ROWS - 1) / GEMM_ROWS;
    const int Mp = mblocks * GEMM_ROWS;

    // ---- workspace carve-up ----
    uint8_t* p = (uint8_t*)d_ws;
    auto take = [&](size_t bytes) -> void* {
        void* r = (void*)p;
        p += (bytes + 255) & ~(size_t)255;
        return r;
    };
    float*  vn     = (float*) take((size_t)M * 3 * sizeof(float));
    __bf16* cubes  = (__bf16*)take((size_t)Mp * 128 * sizeof(__bf16));
    __bf16* zl1    = (__bf16*)take((size_t)Mp * 128 * sizeof(__bf16));
    __bf16* zcat   = (__bf16*)take((size_t)Mp * 256 * sizeof(__bf16));
    __bf16* z2     = (__bf16*)take((size_t)Mp * 512 * sizeof(__bf16));
    __bf16* WconvB = (__bf16*)take((size_t)128 * 128 * sizeof(__bf16));
    __bf16* WlfcB  = (__bf16*)take((size_t)128 * 128 * sizeof(__bf16));
    __bf16* W2B    = (__bf16*)take((size_t)512 * 256 * sizeof(__bf16));
    __bf16* W3B    = (__bf16*)take((size_t)256 * 512 * sizeof(__bf16));

    // ---- weight conversion (f32 -> bf16 [N][K], K padded) ----
    cvt_weight_kernel<<<(128 * 128 + 255) / 256, 256, 0, stream>>>(Wconv, WconvB, 128, 125, 128);
    cvt_weight_kernel<<<(128 * 128 + 255) / 256, 256, 0, stream>>>(Wlfc,  WlfcB,  128, 128, 128);
    cvt_weight_kernel<<<(512 * 256 + 255) / 256, 256, 0, stream>>>(W2,    W2B,    512, 256, 256);
    cvt_weight_kernel<<<(256 * 512 + 255) / 256, 256, 0, stream>>>(W3,    W3B,    256, 512, 512);

    // ---- vertex normals ----
    zero_f32_kernel<<<(M * 3 + 255) / 256, 256, 0, stream>>>(vn, M * 3);
    face_normal_kernel<<<(NF + 255) / 256, 256, 0, stream>>>(v, f, vn, NF);

    // ---- per-vertex sampling + z_point (fills zcat cols 0..127 and cubes,
    //      zero-fills padded rows M..Mp) ----
    vertex_kernel<<<Mp, 128, 0, stream>>>(v, vn, W1, b1, vol, zcat, cubes, M);

    // ---- GEMM chain on WMMA (intermediates unguarded: rows padded to Mp) ----
    // z_local1 = leaky(cubes @ Wconv^T + bconv)            [Mp,128]
    gemm_bf16_kernel<128, false, false><<<mblocks, 256, 0, stream>>>(
        cubes, WconvB, bconv, zl1, Mp, 128, 128, 0);
    // z_local = leaky(z_local1 @ Wlfc^T + blfc) -> zcat cols 128..255
    gemm_bf16_kernel<128, false, false><<<mblocks, 256, 0, stream>>>(
        zl1, WlfcB, blfc, zcat, Mp, 128, 256, 128);
    // z = leaky([z_point, z_local] @ W2^T + b2)            [Mp,512]
    gemm_bf16_kernel<256, false, false><<<mblocks, 256, 0, stream>>>(
        zcat, W2B, b2, z2, Mp, 512, 512, 0);
    // out = leaky(z @ W3^T + b3)                           [M,256] f32, guarded
    gemm_bf16_kernel<512, true, true><<<mblocks, 256, 0, stream>>>(
        z2, W3B, b3, out, M, 256, 256, 0);
}
